// Decoder_33122787787004
// MI455X (gfx1250) — matrix operand order repaired and checked by reading, not compiled, not run on hardware
//
#include <hip/hip_runtime.h>
#include <hip/hip_bf16.h>
#include <math.h>

// Problem sizes
constexpr int Bn = 64, Ln = 512, En = 256, Hn = 256, G4 = 1024; // 4*H

// d_out layout (floats): [att B*L*L][pointers B*L][hT B*H][cT B*H]
constexpr size_t OUT_ATT = 0;
constexpr size_t OUT_PTR = (size_t)Bn * Ln * Ln;            // 16,777,216
constexpr size_t OUT_H   = OUT_PTR + (size_t)Bn * Ln;       // +32,768
constexpr size_t OUT_C   = OUT_H + (size_t)Bn * Hn;         // +16,384

// workspace byte offsets
constexpr size_t OFF_WHH    = 0;                                    // 4H*H bf16 = 512KB
constexpr size_t OFF_WIH    = OFF_WHH  + (size_t)G4 * Hn * 2;       // 512KB
constexpr size_t OFF_WIN    = OFF_WIH  + (size_t)G4 * En * 2;       // 128KB
constexpr size_t OFF_WCTX   = OFF_WIN  + (size_t)Hn * Hn * 2;       // 128KB
constexpr size_t OFF_CTXBF  = OFF_WCTX + (size_t)Hn * Hn * 2;       // 16MB
constexpr size_t OFF_CTXPRJ = OFF_CTXBF + (size_t)Bn * Ln * Hn * 2;  // 32MB f32
constexpr size_t OFF_INP    = OFF_CTXPRJ + (size_t)Bn * Ln * Hn * 4; // 32MB f32

typedef __attribute__((ext_vector_type(16))) __bf16          v16bf;
typedef __attribute__((ext_vector_type(8)))  float           v8f;
typedef __attribute__((ext_vector_type(8)))  unsigned short  ushort8;

union Frag { v16bf bf; ushort8 us[2]; };

static __device__ __forceinline__ unsigned short f2bf(float f) {
    unsigned int u = __float_as_uint(f);
    unsigned int r = u + 0x7FFFu + ((u >> 16) & 1u);   // round-to-nearest-even
    return (unsigned short)(r >> 16);
}

static __device__ __forceinline__ v8f wmma_bf16(const Frag& a, const Frag& b, v8f c) {
    return __builtin_amdgcn_wmma_f32_16x16x32_bf16(false, a.bf, false, b.bf,
                                                   (short)0, c, false, false);
}

// gfx1250 has V_TANH_F32 in the transcendental unit; use it if clang exposes it.
static __device__ __forceinline__ float fast_tanh(float x) {
#if __has_builtin(__builtin_amdgcn_tanhf)
    return __builtin_amdgcn_tanhf(x);
#elif __has_builtin(__builtin_amdgcn_tanh_f32)
    return __builtin_amdgcn_tanh_f32(x);
#else
    float e = __expf(-2.0f * fabsf(x));
    float r = (1.0f - e) / (1.0f + e);
    return copysignf(r, x);
#endif
}

static __device__ __forceinline__ float fast_sigmoid(float x) {
#if __has_builtin(__builtin_amdgcn_rcpf)
    return __builtin_amdgcn_rcpf(1.0f + __expf(-x));
#else
    return 1.0f / (1.0f + __expf(-x));
#endif
}

// ---------------------------------------------------------------- f32 -> bf16
__global__ void cvt_bf16_k(const float* __restrict__ s, unsigned short* __restrict__ d, int n) {
    for (int i = blockIdx.x * blockDim.x + threadIdx.x; i < n; i += gridDim.x * blockDim.x)
        d[i] = f2bf(s[i]);
}

// -------------------------------------------- ctx_proj = context @ Wctx^T + b
// rows = B*L (32768), cols = H (256), K = H (256). One wave per 16x16 tile.
__global__ __launch_bounds__(256) void ctxproj_k(const unsigned short* __restrict__ ctx_bf,
                                                 const unsigned short* __restrict__ wctx,
                                                 const float* __restrict__ b_ctx,
                                                 float* __restrict__ ctx_proj) {
    int wave = threadIdx.x >> 5, lane = threadIdx.x & 31;
    int tile = blockIdx.x * 8 + wave;              // 4096 blocks * 8 waves = 32768 tiles
    int mt = tile >> 4, nt = tile & 15;
    int laneN = lane & 15, laneHi = lane >> 4;

    size_t arow = (size_t)(mt * 16 + laneN);
    size_t bn   = (size_t)(nt * 16 + laneN);
    v8f acc = {};
    for (int k0 = 0; k0 < Hn; k0 += 32) {
        Frag a, b;
        const ushort8* pa = (const ushort8*)(ctx_bf + arow * Hn + k0 + laneHi * 8);
        a.us[0] = pa[0]; a.us[1] = pa[2];          // K chunk and K+16 chunk
        const ushort8* pb = (const ushort8*)(wctx + bn * Hn + k0 + laneHi * 8);
        b.us[0] = pb[0]; b.us[1] = pb[2];
        acc = wmma_bf16(a, b, acc);
    }
    int col = nt * 16 + laneN;
    float bias = b_ctx[col];
#pragma unroll
    for (int r = 0; r < 8; ++r) {
        int row = mt * 16 + laneHi * 8 + r;
        ctx_proj[(size_t)row * Hn + col] = acc[r] + bias;
    }
}

// ------------------------------------------------------- sequential recurrence
// Single workgroup, 1024 threads = 32 waves. h/x in LDS (bf16), c in registers.
__global__ __launch_bounds__(1024) void recur_k(const float* __restrict__ emb,
                                                const float* __restrict__ dec,
                                                const float* __restrict__ h0,
                                                const float* __restrict__ c0,
                                                const unsigned short* __restrict__ Wih,
                                                const float* __restrict__ b_ih,
                                                const unsigned short* __restrict__ Whh,
                                                const float* __restrict__ b_hh,
                                                const unsigned short* __restrict__ Win,
                                                const float* __restrict__ b_in,
                                                float* __restrict__ inp_all,
                                                float* __restrict__ d_out) {
    __shared__ __align__(16) unsigned short x_bf[Bn * En];   // 32 KB
    __shared__ __align__(16) unsigned short h_bf[Bn * Hn];   // 32 KB

    const int tid = threadIdx.x;
    const int wave = tid >> 5, lane = tid & 31;
    const int laneN = lane & 15, laneHi = lane >> 4;
    const int ht = wave & 15;            // h-tile 0..15 (column tile within a gate)
    const int mpair = wave >> 4;         // 0 -> m-tiles {0,1}, 1 -> {2,3}
    const int hcol = ht * 16 + laneN;    // column within H

    // init h
    for (int q = tid; q < Bn * Hn; q += 1024) h_bf[q] = f2bf(h0[q]);

    // per-lane gate biases (i,f,g,o) and inp bias
    float bg[4];
#pragma unroll
    for (int g = 0; g < 4; ++g) bg[g] = b_ih[g * Hn + hcol] + b_hh[g * Hn + hcol];
    const float binv = b_in[hcol];

    // persistent cell state registers: creg[t][r] <-> (row, hcol)
    float creg[2][8];
#pragma unroll
    for (int t = 0; t < 2; ++t) {
        int mt = mpair * 2 + t;
#pragma unroll
        for (int r = 0; r < 8; ++r) {
            int row = mt * 16 + laneHi * 8 + r;
            creg[t][r] = c0[row * Hn + hcol];
        }
    }

    for (int i = 0; i < Ln; ++i) {
        __syncthreads();   // everyone done reading previous x/h
        // stage x_i (bf16) into LDS
        if (i == 0) {
            for (int q = tid; q < Bn * En; q += 1024) x_bf[q] = f2bf(dec[q]);
        } else {
            for (int q = tid; q < Bn * En; q += 1024) {
                int b = q >> 8, e = q & 255;
                x_bf[q] = f2bf(emb[((size_t)b * Ln + (i - 1)) * En + e]);
            }
        }
        __syncthreads();

        // gates = x@Wih^T + h@Whh^T  (this wave: 2 m-tiles x 4 gate tiles)
        v8f acc[2][4] = {};
        for (int g = 0; g < 4; ++g) {
            size_t n = (size_t)(g * Hn + hcol);
            for (int k0 = 0; k0 < Hn; k0 += 32) {
                Frag bih, bhh;
                const ushort8* p1 = (const ushort8*)(Wih + n * En + k0 + laneHi * 8);
                bih.us[0] = p1[0]; bih.us[1] = p1[2];
                const ushort8* p2 = (const ushort8*)(Whh + n * Hn + k0 + laneHi * 8);
                bhh.us[0] = p2[0]; bhh.us[1] = p2[2];
#pragma unroll
                for (int t = 0; t < 2; ++t) {
                    int arow = (mpair * 2 + t) * 16 + laneN;
                    Frag ax, ah;
                    const ushort8* pax = (const ushort8*)(x_bf + arow * En + k0 + laneHi * 8);
                    ax.us[0] = pax[0]; ax.us[1] = pax[2];
                    const ushort8* pah = (const ushort8*)(h_bf + arow * Hn + k0 + laneHi * 8);
                    ah.us[0] = pah[0]; ah.us[1] = pah[2];
                    acc[t][g] = wmma_bf16(ax, bih, acc[t][g]);
                    acc[t][g] = wmma_bf16(ah, bhh, acc[t][g]);
                }
            }
        }
        __syncthreads();   // all h_bf reads done before overwrite

        // LSTM elementwise; write h_t back to LDS (bf16)
#pragma unroll
        for (int t = 0; t < 2; ++t) {
            int mt = mpair * 2 + t;
#pragma unroll
            for (int r = 0; r < 8; ++r) {
                int row = mt * 16 + laneHi * 8 + r;
                float ig = acc[t][0][r] + bg[0];
                float fg = acc[t][1][r] + bg[1];
                float gg = acc[t][2][r] + bg[2];
                float og = acc[t][3][r] + bg[3];
                float cn = fast_sigmoid(fg) * creg[t][r] + fast_sigmoid(ig) * fast_tanh(gg);
                float hn = fast_sigmoid(og) * fast_tanh(cn);
                creg[t][r] = cn;
                h_bf[row * Hn + hcol] = f2bf(hn);
                if (i == Ln - 1) {
                    d_out[OUT_H + (size_t)row * Hn + hcol] = hn;
                    d_out[OUT_C + (size_t)row * Hn + hcol] = cn;
                }
            }
        }
        __syncthreads();

        // inp = h_t @ Win^T + b_in -> inp_all[i][row][col]
#pragma unroll
        for (int t = 0; t < 2; ++t) {
            int mt = mpair * 2 + t;
            v8f a2 = {};
            for (int k0 = 0; k0 < Hn; k0 += 32) {
                Frag ah, bw;
                int arow = mt * 16 + laneN;
                const ushort8* pah = (const ushort8*)(h_bf + arow * Hn + k0 + laneHi * 8);
                ah.us[0] = pah[0]; ah.us[1] = pah[2];
                const ushort8* pbw = (const ushort8*)(Win + (size_t)hcol * Hn + k0 + laneHi * 8);
                bw.us[0] = pbw[0]; bw.us[1] = pbw[2];
                a2 = wmma_bf16(ah, bw, a2);
            }
#pragma unroll
            for (int r = 0; r < 8; ++r) {
                int row = mt * 16 + laneHi * 8 + r;
                inp_all[((size_t)i * Bn + row) * Hn + hcol] = a2[r] + binv;
            }
        }
    }
}

// ------------------------------------------------ attention scores + masking
// block: b = z, 16 i's (x), 32 l's (y). ctx tile transposed in LDS for
// conflict-free reads; inp rows broadcast per wave. float4 staging loads.
__global__ __launch_bounds__(256) void attn_k(const float* __restrict__ ctx_proj,
                                              const float* __restrict__ inp_all,
                                              const float* __restrict__ V,
                                              float* __restrict__ out) {
    __shared__ float ctx_s[Hn * 32];                   // [h][ll]  32 KB
    __shared__ __align__(16) float inp_s[16 * Hn];     // [ii][h]  16 KB
    __shared__ float v_s[Hn];                          //           1 KB
    const int t = threadIdx.x;
    const int b = blockIdx.z, i0 = blockIdx.x * 16, l0 = blockIdx.y * 32;

    // stage ctx tile (32 rows x 256 h) transposed: 2048 float4 loads
#pragma unroll
    for (int j = 0; j < 8; ++j) {
        int q = j * 256 + t;
        int ll = q >> 6, h4 = q & 63;                  // 64 lanes share one ll row
        float4 v = *(const float4*)&ctx_proj[((size_t)b * Ln + l0 + ll) * Hn + h4 * 4];
        ctx_s[(h4 * 4 + 0) * 32 + ll] = v.x;
        ctx_s[(h4 * 4 + 1) * 32 + ll] = v.y;
        ctx_s[(h4 * 4 + 2) * 32 + ll] = v.z;
        ctx_s[(h4 * 4 + 3) * 32 + ll] = v.w;
    }
    // stage inp tile (16 rows x 256 h): 1024 float4 loads/stores
#pragma unroll
    for (int j = 0; j < 4; ++j) {
        int q = j * 256 + t;
        int ii = q >> 6, h4 = q & 63;
        float4 v = *(const float4*)&inp_all[((size_t)(i0 + ii) * Bn + b) * Hn + h4 * 4];
        *(float4*)&inp_s[ii * Hn + h4 * 4] = v;
    }
    v_s[t] = V[t];
    __syncthreads();

    const int ll = t & 31;
    const int iiA = t >> 5;            // 0..7 ; second row iiA+8
    float a0 = 0.f, a1 = 0.f;
#pragma unroll 4
    for (int h = 0; h < Hn; ++h) {
        float vc = v_s[h];
        float cv = ctx_s[h * 32 + ll];
        a0 += vc * fast_tanh(inp_s[iiA * Hn + h] + cv);
        a1 += vc * fast_tanh(inp_s[(iiA + 8) * Hn + h] + cv);
    }
    int l = l0 + ll;
    int i1 = i0 + iiA, i2 = i0 + iiA + 8;
    out[OUT_ATT + ((size_t)b * Ln + i1) * Ln + l] = (l < i1) ? -__builtin_inff() : a0;
    out[OUT_ATT + ((size_t)b * Ln + i2) * Ln + l] = (l < i2) ? -__builtin_inff() : a1;
}

// ----------------------------------------------------------- pointers output
__global__ void ptrs_k(float* __restrict__ out) {
    int q = blockIdx.x * blockDim.x + threadIdx.x;
    if (q < Bn * Ln) out[OUT_PTR + q] = (float)(q & (Ln - 1));
}

extern "C" void kernel_launch(void* const* d_in, const int* in_sizes, int n_in,
                              void* d_out, int out_size, void* d_ws, size_t ws_size,
                              hipStream_t stream) {
    const float* emb  = (const float*)d_in[0];
    const float* dec  = (const float*)d_in[1];
    const float* h0   = (const float*)d_in[2];
    const float* c0   = (const float*)d_in[3];
    const float* ctx  = (const float*)d_in[4];
    const float* Wih  = (const float*)d_in[5];
    const float* bih  = (const float*)d_in[6];
    const float* Whh  = (const float*)d_in[7];
    const float* bhh  = (const float*)d_in[8];
    const float* Win  = (const float*)d_in[9];
    const float* bin  = (const float*)d_in[10];
    const float* Wctx = (const float*)d_in[11];
    const float* bctx = (const float*)d_in[12];
    const float* V    = (const float*)d_in[13];
    float* out = (float*)d_out;

    char* ws = (char*)d_ws;
    unsigned short* whh_bf  = (unsigned short*)(ws + OFF_WHH);
    unsigned short* wih_bf  = (unsigned short*)(ws + OFF_WIH);
    unsigned short* win_bf  = (unsigned short*)(ws + OFF_WIN);
    unsigned short* wctx_bf = (unsigned short*)(ws + OFF_WCTX);
    unsigned short* ctx_bf  = (unsigned short*)(ws + OFF_CTXBF);
    float* ctx_proj = (float*)(ws + OFF_CTXPRJ);
    float* inp_all  = (float*)(ws + OFF_INP);

    cvt_bf16_k<<<512, 256, 0, stream>>>(Whh, whh_bf, G4 * Hn);
    cvt_bf16_k<<<512, 256, 0, stream>>>(Wih, wih_bf, G4 * En);
    cvt_bf16_k<<<256, 256, 0, stream>>>(Win, win_bf, Hn * Hn);
    cvt_bf16_k<<<256, 256, 0, stream>>>(Wctx, wctx_bf, Hn * Hn);
    cvt_bf16_k<<<4096, 256, 0, stream>>>(ctx, ctx_bf, Bn * Ln * Hn);

    ctxproj_k<<<4096, 256, 0, stream>>>(ctx_bf, wctx_bf, bctx, ctx_proj);

    recur_k<<<1, 1024, 0, stream>>>(emb, dec, h0, c0, wih_bf, bih, whh_bf, bhh,
                                    win_bf, bin, inp_all, out);

    attn_k<<<dim3(Ln / 16, Ln / 32, Bn), 256, 0, stream>>>(ctx_proj, inp_all, V, out);

    ptrs_k<<<(Bn * Ln + 255) / 256, 256, 0, stream>>>(out);
}
